// MultiHeadedAttention_3478923509971
// MI455X (gfx1250) — matrix operand order repaired and checked by reading, not compiled, not run on hardware
//
#include <hip/hip_runtime.h>
#include <hip/hip_bf16.h>

// MI455X / gfx1250 multi-head attention, f16 WMMA (f32 accum) + async-LDS staging.
// B=4 S=2048 D=1024 H=16 Dk=64.

#define BB 4
#define SS 2048
#define DM 1024
#define HH 16
#define DKK 64

// LDS B-tile: 64 rows x 32 halves, padded to 40 halves/row (80B) to avoid
// 4-way bank conflicts on the 16-lane fragment reads (stride 20 banks).
#define BROW 40

typedef __attribute__((ext_vector_type(16))) _Float16 v16h;
typedef __attribute__((ext_vector_type(8)))  _Float16 v8h;
typedef __attribute__((ext_vector_type(8)))  float    v8f;

// Async builtin parameter type (from hipcc diagnostic):
//   __attribute__((vector_size(16))) int  in AS(1) (src) / AS(3) (dst)
typedef int v4i_vs __attribute__((__vector_size__(16)));
typedef __attribute__((address_space(1))) v4i_vs* as1_v4p;
typedef __attribute__((address_space(3))) v4i_vs* as3_v4p;

#if __has_builtin(__builtin_amdgcn_global_load_async_to_lds_b128)
#define HAVE_ASYNC 1
#else
#define HAVE_ASYNC 0
#endif

__device__ __forceinline__ v8f wmma16(v16h a, v16h b, v8f c) {
  return __builtin_amdgcn_wmma_f32_16x16x32_f16(
      /*neg_a=*/false, a, /*neg_b=*/false, b,
      /*c_mod=*/(short)0, c, /*reuse_a=*/false, /*reuse_b=*/false);
}

// Stage one 64x32-half W tile (rows n0..n0+63, cols k0..k0+31) into LDS.
// 256 threads x 16B each. gsrc = Wh + n0*DM + k0.
__device__ __forceinline__ void stage_issue(_Float16* ldsdst,
                                            const _Float16* gsrc, int tid) {
  const int r = tid >> 2;
  const int c = (tid & 3) * 8;
#if HAVE_ASYNC
  __builtin_amdgcn_global_load_async_to_lds_b128(
      (as1_v4p)(_Float16*)(gsrc + (size_t)r * DM + c),
      (as3_v4p)(ldsdst + r * BROW + c),
      /*offset=*/0, /*cpol=*/0);
#else
  v8h tmp = *(const v8h*)(gsrc + (size_t)r * DM + c);
  *(v8h*)(ldsdst + r * BROW + c) = tmp;
#endif
}

__device__ __forceinline__ void stage_wait() {
#if HAVE_ASYNC
  asm volatile("s_wait_asynccnt 0x0" ::: "memory");
#else
  asm volatile("s_wait_dscnt 0x0" ::: "memory");
#endif
}

// Read B fragment t from staged LDS tile: column n-local = t*16+l16, K = 8g..8g+7, 16+8g..16+8g+7.
__device__ __forceinline__ v16h ldsB(const _Float16* tile, int t, int l16, int g) {
  const _Float16* p = tile + (t * 16 + l16) * BROW + 8 * g;
  v8h b0 = *(const v8h*)(p);
  v8h b1 = *(const v8h*)(p + 16);
  v16h b;
#pragma unroll
  for (int i = 0; i < 8; ++i) { b[i] = b0[i]; b[i + 8] = b1[i]; }
  return b;
}

// A fragment from f32 row (converted on the fly). ap = row + k0 + 8g.
__device__ __forceinline__ v16h loadA_f32(const float* ap) {
  float4 f0 = *(const float4*)(ap);
  float4 f1 = *(const float4*)(ap + 4);
  float4 f2 = *(const float4*)(ap + 16);
  float4 f3 = *(const float4*)(ap + 20);
  v16h a;
  a[0]=(_Float16)f0.x;  a[1]=(_Float16)f0.y;  a[2]=(_Float16)f0.z;  a[3]=(_Float16)f0.w;
  a[4]=(_Float16)f1.x;  a[5]=(_Float16)f1.y;  a[6]=(_Float16)f1.z;  a[7]=(_Float16)f1.w;
  a[8]=(_Float16)f2.x;  a[9]=(_Float16)f2.y;  a[10]=(_Float16)f2.z; a[11]=(_Float16)f2.w;
  a[12]=(_Float16)f3.x; a[13]=(_Float16)f3.y; a[14]=(_Float16)f3.z; a[15]=(_Float16)f3.w;
  return a;
}

__device__ __forceinline__ v16h loadA_f16(const _Float16* ap) {
  v8h a0 = *(const v8h*)(ap);
  v8h a1 = *(const v8h*)(ap + 16);
  v16h a;
#pragma unroll
  for (int i = 0; i < 8; ++i) { a[i] = a0[i]; a[i + 8] = a1[i]; }
  return a;
}

// ---------------------------------------------------------------- cvt f32->f16
__global__ __launch_bounds__(256) void cvt_w_kernel(const float* __restrict__ src,
                                                    _Float16* __restrict__ dst, int n) {
  int i = (blockIdx.x * 256 + threadIdx.x) * 4;
  if (i + 3 < n) {
    float4 v = *(const float4*)(src + i);
    dst[i + 0] = (_Float16)v.x;
    dst[i + 1] = (_Float16)v.y;
    dst[i + 2] = (_Float16)v.z;
    dst[i + 3] = (_Float16)v.w;
  }
}

// ------------------------------------------------------- QKV projection GEMM
// Block tile 256(M) x 64(N); wave tile 32(M) x 64(N); W staged via async LDS.
// mode 0: Q -> [B,H,S,Dk] f16 scaled by 1/8; 1: K -> [B,H,S,Dk]; 2: V -> [B,H,Dk,S].
__global__ __launch_bounds__(256) void proj_qkv_kernel(const float* __restrict__ X,
                                                       const _Float16* __restrict__ Wh,
                                                       const float* __restrict__ bias,
                                                       _Float16* __restrict__ Out,
                                                       int mode) {
  __shared__ __align__(64) _Float16 Bt[2][64 * BROW];

  const int tid  = threadIdx.x;
  const int wave = tid >> 5;
  const int lane = tid & 31;
  const int g    = lane >> 4;
  const int l16  = lane & 15;
  const int mb   = blockIdx.x >> 4;   // 32 M-blocks of 256 rows
  const int nb   = blockIdx.x & 15;   // 16 N-blocks of 64 cols
  const int n0   = nb * 64;
  const int m0   = mb * 256 + wave * 32;

  v8f acc[2][4];
#pragma unroll
  for (int s = 0; s < 2; ++s)
#pragma unroll
    for (int t = 0; t < 4; ++t) acc[s][t] = 0;

  const float* arow0 = X + (size_t)(m0 + l16) * DM + 8 * g;
  const float* arow1 = arow0 + (size_t)16 * DM;
  const _Float16* wsrc = Wh + (size_t)n0 * DM;

  stage_issue(&Bt[0][0], wsrc, tid);
  int cur = 0;

  for (int k0 = 0; k0 < DM; k0 += 32) {
    stage_wait();
    __syncthreads();
    if (k0 + 32 < DM) stage_issue(&Bt[cur ^ 1][0], wsrc + k0 + 32, tid);

    v16h a0 = loadA_f32(arow0 + k0);
    v16h a1 = loadA_f32(arow1 + k0);
    v16h b0 = ldsB(&Bt[cur][0], 0, l16, g);
    v16h b1 = ldsB(&Bt[cur][0], 1, l16, g);
    v16h b2 = ldsB(&Bt[cur][0], 2, l16, g);
    v16h b3 = ldsB(&Bt[cur][0], 3, l16, g);

    acc[0][0] = wmma16(a0, b0, acc[0][0]);
    acc[1][0] = wmma16(a1, b0, acc[1][0]);
    acc[0][1] = wmma16(a0, b1, acc[0][1]);
    acc[1][1] = wmma16(a1, b1, acc[1][1]);
    acc[0][2] = wmma16(a0, b2, acc[0][2]);
    acc[1][2] = wmma16(a1, b2, acc[1][2]);
    acc[0][3] = wmma16(a0, b3, acc[0][3]);
    acc[1][3] = wmma16(a1, b3, acc[1][3]);
    cur ^= 1;
  }

#pragma unroll
  for (int s = 0; s < 2; ++s) {
#pragma unroll
    for (int t = 0; t < 4; ++t) {
      const int col = n0 + t * 16 + l16;
      const float bv = bias[col];
      const int h = col >> 6, d = col & (DKK - 1);
#pragma unroll
      for (int e = 0; e < 8; ++e) {
        const int row = m0 + s * 16 + e + 8 * g;
        const int bi = row >> 11, si = row & (SS - 1);
        float v = acc[s][t][e] + bv;
        if (mode == 0) v *= 0.125f;  // 1/sqrt(Dk)
        size_t idx;
        if (mode == 2) idx = ((size_t)(bi * HH + h) * DKK + d) * SS + si;
        else           idx = ((size_t)(bi * HH + h) * SS + si) * DKK + d;
        Out[idx] = (_Float16)v;
      }
    }
  }
}

// --------------------------------------------------------- flash attention
__global__ __launch_bounds__(256) void attn_kernel(const _Float16* __restrict__ Q,
                                                   const _Float16* __restrict__ K,
                                                   const _Float16* __restrict__ Vt,
                                                   const int* __restrict__ mask,
                                                   _Float16* __restrict__ Xatt) {
  __shared__ __align__(64) _Float16 plds[8][16 * 32];

  const int tid  = threadIdx.x;
  const int wave = tid >> 5;
  const int lane = tid & 31;
  const int g    = lane >> 4;
  const int l16  = lane & 15;
  const int wt   = blockIdx.x * 8 + wave;
  const int qt   = wt & 127;
  const int bh   = wt >> 7;
  const int bi   = bh >> 4;
  const int h    = bh & 15;
  const int q0   = qt * 16;

  const _Float16* Qp = Q  + ((size_t)bh * SS + q0) * DKK;
  const _Float16* Kp = K  + (size_t)bh * SS * DKK;
  const _Float16* Vp = Vt + (size_t)bh * DKK * SS;
  const int*      mp = mask + ((size_t)bi * SS + q0) * SS;

  const _Float16* qr = Qp + (size_t)l16 * DKK;
  v8h q00 = *(const v8h*)(qr + 16 * g);
  v8h q01 = *(const v8h*)(qr + 8 + 16 * g);
  v8h q10 = *(const v8h*)(qr + 32 + 16 * g);
  v8h q11 = *(const v8h*)(qr + 40 + 16 * g);
  v16h aq0, aq1;
#pragma unroll
  for (int i = 0; i < 8; ++i) {
    aq0[i] = q00[i]; aq0[i + 8] = q01[i];
    aq1[i] = q10[i]; aq1[i + 8] = q11[i];
  }

  float m_r[8], l_r[8];
#pragma unroll
  for (int e = 0; e < 8; ++e) { m_r[e] = -3.0e38f; l_r[e] = 0.0f; }
  v8f o[4];
#pragma unroll
  for (int t = 0; t < 4; ++t) o[t] = 0;

  _Float16* lds = &plds[wave][0];

  for (int kb = 0; kb < SS / 32; ++kb) {
    const int j0 = kb * 32;

    const _Float16* kr0 = Kp + (size_t)(j0 + l16) * DKK + 16 * g;
    const _Float16* kr1 = kr0 + (size_t)16 * DKK;
    // prefetch next key/value block while this one computes
    __builtin_prefetch((const void*)(kr0 + (size_t)32 * DKK), 0, 0);
    __builtin_prefetch((const void*)(Vp + (size_t)l16 * SS + j0 + 32), 0, 0);

    v16h bk00 = *(const v16h*)(kr0);
    v16h bk01 = *(const v16h*)(kr0 + 32);
    v16h bk10 = *(const v16h*)(kr1);
    v16h bk11 = *(const v16h*)(kr1 + 32);
    v8f s0 = 0, s1 = 0;
    s0 = wmma16(aq0, bk00, s0);
    s0 = wmma16(aq1, bk01, s0);
    s1 = wmma16(aq0, bk10, s1);
    s1 = wmma16(aq1, bk11, s1);

#pragma unroll
    for (int e = 0; e < 8; ++e) {
      const int* mr = mp + (size_t)(e + 8 * g) * SS + j0 + l16;
      float v0 = mr[0]  ? s0[e] : -1e9f;
      float v1 = mr[16] ? s1[e] : -1e9f;
      float mx = fmaxf(v0, v1);
#pragma unroll
      for (int off = 1; off < 16; off <<= 1) mx = fmaxf(mx, __shfl_xor(mx, off, 32));
      const float mnew  = fmaxf(m_r[e], mx);
      const float alpha = __expf(m_r[e] - mnew);
      const float p0 = __expf(v0 - mnew);
      const float p1 = __expf(v1 - mnew);
      float rs = p0 + p1;
#pragma unroll
      for (int off = 1; off < 16; off <<= 1) rs += __shfl_xor(rs, off, 32);
      l_r[e] = l_r[e] * alpha + rs;
      m_r[e] = mnew;
      o[0][e] *= alpha; o[1][e] *= alpha; o[2][e] *= alpha; o[3][e] *= alpha;
      lds[(e + 8 * g) * 32 + l16]      = (_Float16)p0;
      lds[(e + 8 * g) * 32 + 16 + l16] = (_Float16)p1;
    }
    asm volatile("s_wait_dscnt 0x0" ::: "memory");

    const _Float16* pr = lds + (size_t)l16 * 32 + 16 * g;
    v8h p0v = *(const v8h*)(pr);
    v8h p1v = *(const v8h*)(pr + 8);
    v16h ap;
#pragma unroll
    for (int i = 0; i < 8; ++i) { ap[i] = p0v[i]; ap[i + 8] = p1v[i]; }

#pragma unroll
    for (int t = 0; t < 4; ++t) {
      v16h bv = *(const v16h*)(Vp + (size_t)(t * 16 + l16) * SS + j0 + 16 * g);
      o[t] = wmma16(ap, bv, o[t]);
    }
  }

#pragma unroll
  for (int t = 0; t < 4; ++t) {
#pragma unroll
    for (int e = 0; e < 8; ++e) {
      const int srow = q0 + e + 8 * g;
      const float v = o[t][e] / l_r[e];
      Xatt[((size_t)bi * SS + srow) * DM + h * DKK + t * 16 + l16] = (_Float16)v;
    }
  }
}

// ------------------------------------------------------ output projection GEMM
__global__ __launch_bounds__(256) void oproj_kernel(const _Float16* __restrict__ X,
                                                    const _Float16* __restrict__ Wh,
                                                    const float* __restrict__ bias,
                                                    float* __restrict__ Out) {
  __shared__ __align__(64) _Float16 Bt[2][64 * BROW];

  const int tid  = threadIdx.x;
  const int wave = tid >> 5;
  const int lane = tid & 31;
  const int g    = lane >> 4;
  const int l16  = lane & 15;
  const int mb   = blockIdx.x >> 4;
  const int nb   = blockIdx.x & 15;
  const int n0   = nb * 64;
  const int m0   = mb * 256 + wave * 32;

  v8f acc[2][4];
#pragma unroll
  for (int s = 0; s < 2; ++s)
#pragma unroll
    for (int t = 0; t < 4; ++t) acc[s][t] = 0;

  const _Float16* arow0 = X + (size_t)(m0 + l16) * DM + 8 * g;
  const _Float16* arow1 = arow0 + (size_t)16 * DM;
  const _Float16* wsrc  = Wh + (size_t)n0 * DM;

  stage_issue(&Bt[0][0], wsrc, tid);
  int cur = 0;

  for (int k0 = 0; k0 < DM; k0 += 32) {
    stage_wait();
    __syncthreads();
    if (k0 + 32 < DM) stage_issue(&Bt[cur ^ 1][0], wsrc + k0 + 32, tid);

    v16h a0 = loadA_f16(arow0 + k0);
    v16h a1 = loadA_f16(arow1 + k0);
    v16h b0 = ldsB(&Bt[cur][0], 0, l16, g);
    v16h b1 = ldsB(&Bt[cur][0], 1, l16, g);
    v16h b2 = ldsB(&Bt[cur][0], 2, l16, g);
    v16h b3 = ldsB(&Bt[cur][0], 3, l16, g);

    acc[0][0] = wmma16(a0, b0, acc[0][0]);
    acc[1][0] = wmma16(a1, b0, acc[1][0]);
    acc[0][1] = wmma16(a0, b1, acc[0][1]);
    acc[1][1] = wmma16(a1, b1, acc[1][1]);
    acc[0][2] = wmma16(a0, b2, acc[0][2]);
    acc[1][2] = wmma16(a1, b2, acc[1][2]);
    acc[0][3] = wmma16(a0, b3, acc[0][3]);
    acc[1][3] = wmma16(a1, b3, acc[1][3]);
    cur ^= 1;
  }

#pragma unroll
  for (int s = 0; s < 2; ++s) {
#pragma unroll
    for (int t = 0; t < 4; ++t) {
      const int col = n0 + t * 16 + l16;
      const float bv = bias[col];
#pragma unroll
      for (int e = 0; e < 8; ++e) {
        const int row = m0 + s * 16 + e + 8 * g;
        Out[(size_t)row * DM + col] = acc[s][t][e] + bv;
      }
    }
  }
}

// ---------------------------------------------------------------------- launch
extern "C" void kernel_launch(void* const* d_in, const int* in_sizes, int n_in,
                              void* d_out, int out_size, void* d_ws, size_t ws_size,
                              hipStream_t stream) {
  const float* query  = (const float*)d_in[0];
  const float* key_in = (const float*)d_in[1];
  const float* value  = (const float*)d_in[2];
  const int*   mask   = (const int*)d_in[3];
  const float* Wq = (const float*)d_in[4];
  const float* bq = (const float*)d_in[5];
  const float* Wk = (const float*)d_in[6];
  const float* bk = (const float*)d_in[7];
  const float* Wv = (const float*)d_in[8];
  const float* bv = (const float*)d_in[9];
  const float* Wo = (const float*)d_in[10];
  const float* bo = (const float*)d_in[11];

  _Float16* ws = (_Float16*)d_ws;
  const size_t WN = (size_t)DM * DM;
  const size_t QN = (size_t)BB * HH * SS * DKK;
  _Float16* Wqh  = ws;
  _Float16* Wkh  = Wqh + WN;
  _Float16* Wvh  = Wkh + WN;
  _Float16* Woh  = Wvh + WN;
  _Float16* Qh   = Woh + WN;
  _Float16* Kh   = Qh + QN;
  _Float16* Vth  = Kh + QN;
  _Float16* Xatt = Vth + QN;

  dim3 blk(256);

  cvt_w_kernel<<<(int)(WN / 1024), blk, 0, stream>>>(Wq, Wqh, (int)WN);
  cvt_w_kernel<<<(int)(WN / 1024), blk, 0, stream>>>(Wk, Wkh, (int)WN);
  cvt_w_kernel<<<(int)(WN / 1024), blk, 0, stream>>>(Wv, Wvh, (int)WN);
  cvt_w_kernel<<<(int)(WN / 1024), blk, 0, stream>>>(Wo, Woh, (int)WN);

  proj_qkv_kernel<<<512, blk, 0, stream>>>(query,  Wqh, bq, Qh,  0);
  proj_qkv_kernel<<<512, blk, 0, stream>>>(key_in, Wkh, bk, Kh,  1);
  proj_qkv_kernel<<<512, blk, 0, stream>>>(value,  Wvh, bv, Vth, 2);

  attn_kernel<<<1024, blk, 0, stream>>>(Qh, Kh, Vth, mask, Xatt);

  oproj_kernel<<<512, blk, 0, stream>>>(Xatt, Woh, bo, (float*)d_out);
}
